// CombinedLoss_25804163514703
// MI455X (gfx1250) — matrix operand order repaired and checked by reading, not compiled, not run on hardware
//
#include <hip/hip_runtime.h>

#define BATCH 4096
#define EMB   512
#define NCLS  1000
#define MARGIN 0.5f
#define ALPHA  0.7f
#define BETA   0.1f

typedef __attribute__((ext_vector_type(16))) __bf16 v16bf;
typedef __attribute__((ext_vector_type(8)))  float  v8f;
typedef __attribute__((ext_vector_type(4)))  unsigned int v4u;

union BfFrag { v16bf bf; v4u u[2]; };

#define INF_F __builtin_inff()

// ---------------------------------------------------------------------------
// Init: sentinels for hardest-pos/neg + zero the scalar accumulators.
// ---------------------------------------------------------------------------
__global__ void init_kernel(float* hp, float* hn, float* scal) {
    int i = blockIdx.x * blockDim.x + threadIdx.x;
    if (i < BATCH) { hp[i] = -INF_F; hn[i] = INF_F; }
    if (i < 2) scal[i] = 0.0f;
}

// ---------------------------------------------------------------------------
// Prep: fp32 -> bf16 convert (RNE), row squared norms, center-loss partial.
// One 256-thread block per embedding row.
// ---------------------------------------------------------------------------
__global__ __launch_bounds__(256)
void prep_kernel(const float* __restrict__ emb, const int* __restrict__ labels,
                 const float* __restrict__ centers, unsigned short* __restrict__ ebf,
                 float* __restrict__ sq, float* __restrict__ center_sum) {
    __shared__ float s1[256], s2[256];
    const int row = blockIdx.x;
    const int t = threadIdx.x;
    const int lab = labels[row];
    const float* e = emb + (size_t)row * EMB;
    const float* c = centers + (size_t)lab * EMB;
    float se = 0.0f, sc = 0.0f;
    for (int j = t; j < EMB; j += 256) {
        float v = e[j];
        se += v * v;
        float d = v - c[j];
        sc += d * d;
        unsigned int u = __float_as_uint(v);
        u += 0x7FFFu + ((u >> 16) & 1u);          // round-to-nearest-even bf16
        ebf[(size_t)row * EMB + j] = (unsigned short)(u >> 16);
    }
    s1[t] = se; s2[t] = sc;
    __syncthreads();
    for (int s = 128; s > 0; s >>= 1) {
        if (t < s) { s1[t] += s1[t + s]; s2[t] += s2[t + s]; }
        __syncthreads();
    }
    if (t == 0) { sq[row] = s1[0]; atomicAdd(center_sum, s2[0]); }
}

// ---------------------------------------------------------------------------
// Gram + distance + hardest pos/neg.  Each wave: one 16-row tile (I) crossed
// with 16 column tiles (J group); K loop in steps of 32 with
// v_wmma_f32_16x16x32_bf16.  Tile is reduced in-register; 4-step shuffle
// collapses 16 columns; one atomicMax/atomicMin per row per wave.
// ---------------------------------------------------------------------------
__global__ __launch_bounds__(256)
void gram_kernel(const unsigned short* __restrict__ ebf, const float* __restrict__ sq,
                 const int* __restrict__ labels, float* __restrict__ hp,
                 float* __restrict__ hn) {
    const int lane = threadIdx.x & 31;
    const int wave = threadIdx.x >> 5;
    const int task = blockIdx.x * 8 + wave;   // 0..4095
    const int I  = task >> 4;                 // row tile     0..255
    const int Jg = task & 15;                 // column group 0..15

    const int half = lane >> 4;               // 0: lanes 0-15, 1: lanes 16-31
    const int l15  = lane & 15;

    // A fragment source row (ISA 16-bit A 16x32 layout): lane holds row I*16+l15,
    // elements 0..7 = K[half*8 .. +7], elements 8..15 = K[16+half*8 .. +7].
    const unsigned short* Abase = ebf + (size_t)(I * 16 + l15) * EMB;
    const int ka = half * 8;
    const int kb = half * 16;                 // unused: B (row J*16+l15) is loaded with ka, same K order as A

    // Per-element output rows (C layout: VGPR r -> M = r + half*8).
    float rsq[8]; int rlab[8];
#pragma unroll
    for (int r = 0; r < 8; ++r) {
        int grow = I * 16 + r + half * 8;
        rsq[r] = sq[grow];
        rlab[r] = labels[grow];
    }

    float hpk[8], hnk[8];
#pragma unroll
    for (int r = 0; r < 8; ++r) { hpk[r] = -INF_F; hnk[r] = INF_F; }

    for (int jj = 0; jj < 16; ++jj) {
        const int J = Jg * 16 + jj;
        const int brow = J * 16 + l15;
        const unsigned short* Bbase = ebf + (size_t)brow * EMB;
        const float csq  = sq[brow];
        const int   clab = labels[brow];

        v8f acc = {};
        for (int k = 0; k < EMB; k += 32) {
            BfFrag a, b;
            a.u[0] = *(const v4u*)(Abase + k + ka);
            a.u[1] = *(const v4u*)(Abase + k + 16 + ka);
            b.u[0] = *(const v4u*)(Bbase + k + ka);
            b.u[1] = *(const v4u*)(Bbase + k + 16 + ka);
            acc = __builtin_amdgcn_wmma_f32_16x16x32_bf16(
                false, a.bf, false, b.bf, (short)0, acc, false, false);
        }

#pragma unroll
        for (int r = 0; r < 8; ++r) {
            const int grow = I * 16 + r + half * 8;
            const int gcol = J * 16 + l15;
            float d2 = rsq[r] + csq - 2.0f * acc[r];
            float dist = sqrtf(fmaxf(d2, 1e-12f));
            bool same = (rlab[r] == clab);
            if (same && (grow != gcol)) hpk[r] = fmaxf(hpk[r], dist);
            if (!same)                  hnk[r] = fminf(hnk[r], dist);
        }
    }

    // Collapse the 16 columns held by each 16-lane half, then one atomic/row.
#pragma unroll
    for (int r = 0; r < 8; ++r) {
        float mx = hpk[r], mn = hnk[r];
#pragma unroll
        for (int m = 1; m < 16; m <<= 1) {
            mx = fmaxf(mx, __shfl_xor(mx, m, 32));
            mn = fminf(mn, __shfl_xor(mn, m, 32));
        }
        if (l15 == 0) {
            const int grow = I * 16 + r + half * 8;
            if (mx > -INF_F) atomicMax(&hp[grow], mx);
            if (mn <  INF_F) atomicMin(&hn[grow], mn);
        }
    }
}

// ---------------------------------------------------------------------------
// Cross-entropy: one 256-thread block per row of logits.
// ---------------------------------------------------------------------------
__global__ __launch_bounds__(256)
void ce_kernel(const float* __restrict__ logits, const int* __restrict__ labels,
               float* __restrict__ ce_sum) {
    __shared__ float sh[256];
    const int row = blockIdx.x;
    const int t = threadIdx.x;
    const float* L = logits + (size_t)row * NCLS;

    float mx = -INF_F;
    for (int j = t; j < NCLS; j += 256) mx = fmaxf(mx, L[j]);
    sh[t] = mx;
    __syncthreads();
    for (int s = 128; s > 0; s >>= 1) {
        if (t < s) sh[t] = fmaxf(sh[t], sh[t + s]);
        __syncthreads();
    }
    const float m = sh[0];
    __syncthreads();

    float se = 0.0f;
    for (int j = t; j < NCLS; j += 256) se += expf(L[j] - m);
    sh[t] = se;
    __syncthreads();
    for (int s = 128; s > 0; s >>= 1) {
        if (t < s) sh[t] += sh[t + s];
        __syncthreads();
    }
    if (t == 0) {
        float logp = L[labels[row]] - m - logf(sh[0]);
        atomicAdd(ce_sum, -logp);
    }
}

// ---------------------------------------------------------------------------
// Finalize: triplet per-row ReLU + validity, combine all three losses.
// ---------------------------------------------------------------------------
__global__ __launch_bounds__(256)
void finalize_kernel(const float* __restrict__ hp, const float* __restrict__ hn,
                     const float* __restrict__ scal, float* __restrict__ out) {
    __shared__ float ssum[256];
    __shared__ int   scnt[256];
    const int t = threadIdx.x;
    float s = 0.0f; int c = 0;
    for (int i = t; i < BATCH; i += 256) {
        float p = hp[i], n = hn[i];
        if (p > -INF_F && n < INF_F) {
            float v = p - n + MARGIN;
            s += (v > 0.0f) ? v : 0.0f;
            c += 1;
        }
    }
    ssum[t] = s; scnt[t] = c;
    __syncthreads();
    for (int st = 128; st > 0; st >>= 1) {
        if (t < st) { ssum[t] += ssum[t + st]; scnt[t] += scnt[t + st]; }
        __syncthreads();
    }
    if (t == 0) {
        int nv = scnt[0] > 1 ? scnt[0] : 1;
        float trip   = ssum[0] / (float)nv;
        float ce     = scal[0] / (float)BATCH;
        float center = scal[1] / (float)(BATCH * EMB);
        float total  = (1.0f - ALPHA - BETA) * ce + ALPHA * trip + BETA * center;
        out[0] = total; out[1] = ce; out[2] = trip;
    }
}

// ---------------------------------------------------------------------------
extern "C" void kernel_launch(void* const* d_in, const int* in_sizes, int n_in,
                              void* d_out, int out_size, void* d_ws, size_t ws_size,
                              hipStream_t stream) {
    const float* emb     = (const float*)d_in[0];
    const float* logits  = (const float*)d_in[1];
    const int*   labels  = (const int*)d_in[2];
    const float* centers = (const float*)d_in[3];
    float* out = (float*)d_out;

    char* ws = (char*)d_ws;
    unsigned short* ebf = (unsigned short*)ws;                 // 4096*512*2 = 4 MB
    float* sq   = (float*)(ws + (size_t)BATCH * EMB * 2);      // 4096 floats
    float* hp   = sq + BATCH;                                  // 4096 floats
    float* hn   = hp + BATCH;                                  // 4096 floats
    float* scal = hn + BATCH;                                  // [0]=ce_sum [1]=center_sum

    init_kernel<<<(BATCH + 255) / 256, 256, 0, stream>>>(hp, hn, scal);
    prep_kernel<<<BATCH, 256, 0, stream>>>(emb, labels, centers, ebf, sq, scal + 1);
    gram_kernel<<<512, 256, 0, stream>>>(ebf, sq, labels, hp, hn);
    ce_kernel<<<BATCH, 256, 0, stream>>>(logits, labels, scal);
    finalize_kernel<<<1, 256, 0, stream>>>(hp, hn, scal, out);
}
